// NF4LinearWrapper_10462540333702
// MI455X (gfx1250) — matrix operand order repaired and checked by reading, not compiled, not run on hardware
//
#include <hip/hip_runtime.h>
#include <hip/hip_bf16.h>

// ---------------------------------------------------------------------------
// NF4 linear: out[16384,4096] = x[16384,4096] @ dequant_nf4(W[4096,4096]).T
// Plan: (1) NF4 fake-quant-dequant W -> f16 in workspace
//       (2) convert x f32 -> f16 in workspace
//       (3) f16 WMMA GEMM with f32 accumulation (v_wmma_f32_16x16x32_f16)
//           wave tile 64x64 -> 16 WMMA per 16 b128 loads per K-step
// ---------------------------------------------------------------------------

typedef __attribute__((ext_vector_type(16))) _Float16 v16h;
typedef __attribute__((ext_vector_type(8)))  _Float16 v8h;
typedef __attribute__((ext_vector_type(4)))  _Float16 v4h;
typedef __attribute__((ext_vector_type(8)))  float    v8f;

#define DIM_M 16384
#define DIM_N 4096
#define DIM_K 4096

// ---------------- NF4 codebook (QLoRA) + midpoint boundaries ----------------
__device__ __constant__ float NF4_CODE[16] = {
    -1.0f, -0.6961928009986877f, -0.5250730514526367f, -0.39491748809814453f,
    -0.28444138169288635f, -0.18477343022823334f, -0.09105003625154495f, 0.0f,
    0.07958029955625534f, 0.16093020141124725f, 0.24611230194568634f,
    0.33791524171829224f, 0.44070982933044434f, 0.5626170039176941f,
    0.7229568362236023f, 1.0f};

__device__ __constant__ float NF4_BOUND[15] = {
    -0.8480964004993439f, -0.6106329262256622f, -0.4599952697753906f,
    -0.3396794348955154f, -0.2346074059605598f, -0.1379117332398891f,
    -0.0455250181257725f,  0.0397901497781277f,  0.1202552504837513f,
     0.2035212516784668f,  0.2920137718319893f,  0.3893125355243683f,
     0.5016634166240692f,  0.6427869200706482f,  0.8614784181118011f};

// ---------------------------------------------------------------------------
// Kernel 1: NF4 quant->dequant of weight, emitting f16.
// One 256-thread workgroup handles one scale-block (256 NF4 blocks of 64).
// ---------------------------------------------------------------------------
__global__ __launch_bounds__(256)
void nf4_dequant_kernel(const float* __restrict__ w, _Float16* __restrict__ wh) {
    __shared__ float red[256];
    const int t = threadIdx.x;
    const size_t base = (((size_t)blockIdx.x * 256) + t) * 64;

    // pass 1: per-NF4-block absmax (each thread owns one 64-elem block)
    const float4* w4 = (const float4*)(w + base);
    float amax = 0.0f;
#pragma unroll
    for (int i = 0; i < 16; ++i) {
        float4 v = w4[i];
        amax = fmaxf(amax, fabsf(v.x));
        amax = fmaxf(amax, fabsf(v.y));
        amax = fmaxf(amax, fabsf(v.z));
        amax = fmaxf(amax, fabsf(v.w));
    }

    // workgroup max-reduce over the 256 absmax values -> scale-block amax
    red[t] = amax;
    __syncthreads();
#pragma unroll
    for (int s = 128; s > 0; s >>= 1) {
        if (t < s) red[t] = fmaxf(red[t], red[t + s]);
        __syncthreads();
    }
    float s_amax = red[0];
    s_amax = (s_amax == 0.0f) ? 1.0f : s_amax;

    // double-quantize this thread's block scale to 8 bits (symmetric)
    float q8 = rintf(amax / s_amax * 127.0f);
    q8 = fminf(fmaxf(q8, -127.0f), 127.0f);
    const float scale_dq   = q8 * (1.0f / 127.0f) * s_amax;
    const float inv_absmax = 1.0f / ((amax == 0.0f) ? 1.0f : amax);

    // pass 2: quantize each element to nearest NF4 code, dequantize, store f16
    v4h* out4 = (v4h*)(wh + base);
#pragma unroll
    for (int i = 0; i < 16; ++i) {
        float4 v = w4[i];
        float e[4] = {v.x, v.y, v.z, v.w};
        v4h h;
#pragma unroll
        for (int c = 0; c < 4; ++c) {
            float n = e[c] * inv_absmax;
            int idx = 0;
#pragma unroll
            for (int b = 0; b < 15; ++b) idx += (n > NF4_BOUND[b]) ? 1 : 0;
            h[c] = (_Float16)(NF4_CODE[idx] * scale_dq);
        }
        out4[i] = h;
    }
}

// ---------------------------------------------------------------------------
// Kernel 2: x f32 -> f16 streaming convert (float4 in, half4 out)
// ---------------------------------------------------------------------------
__global__ __launch_bounds__(256)
void f32_to_f16_kernel(const float* __restrict__ in, _Float16* __restrict__ out,
                       long n4) {
    const long stride = (long)gridDim.x * blockDim.x;
    for (long i = (long)blockIdx.x * blockDim.x + threadIdx.x; i < n4; i += stride) {
        float4 v = ((const float4*)in)[i];
        v4h h;
        h[0] = (_Float16)v.x; h[1] = (_Float16)v.y;
        h[2] = (_Float16)v.z; h[3] = (_Float16)v.w;
        ((v4h*)out)[i] = h;
    }
}

// ---------------------------------------------------------------------------
// Kernel 3: f16 WMMA GEMM, f32 accumulate.
//   out[m,n] = sum_k xh[m,k] * wh[n,k]   (both contiguous along k)
// 256 threads = 8 waves; wave grid 2(M)x4(N); wave tile 64x64; WG tile 128x256.
// K-step 32 -> 16 independent v_wmma_f32_16x16x32_f16 per step.
//
// Fragment addressing follows the CDNA5 VGPR layouts (ISA 7.12.2):
//  A 16x32 f16: lane L (h=L>>4, r=L&15) holds K = {k0+8h .. +7} u {k0+16+8h .. +7}
//  B 32x16 f16: lane L holds N=r, K = {k0+8h .. +7} u {k0+16+8h .. +7}
// ---------------------------------------------------------------------------
__device__ __forceinline__ v16h load_frag(const _Float16* p0, const _Float16* p1) {
    union { v16h v; v8h h[2]; } u;
    u.h[0] = *(const v8h*)p0;   // global_load_b128
    u.h[1] = *(const v8h*)p1;   // global_load_b128
    return u.v;
}

__global__ __launch_bounds__(256)
void gemm_f16_wmma_kernel(const _Float16* __restrict__ xh,
                          const _Float16* __restrict__ wh,
                          float* __restrict__ out) {
    const int lane = threadIdx.x & 31;
    const int wv   = threadIdx.x >> 5;     // wave id 0..7 (wave32)
    const int wm   = wv >> 2;              // 0..1  (M direction)
    const int wn   = wv & 3;               // 0..3  (N direction)
    const int r16  = lane & 15;
    const int half = lane >> 4;

    const size_t mBase = (size_t)blockIdx.y * 128 + (size_t)wm * 64;
    const size_t nBase = (size_t)blockIdx.x * 256 + (size_t)wn * 64;

    // per-lane base pointers (K offset added in the loop)
    const _Float16* aP[4];
#pragma unroll
    for (int i = 0; i < 4; ++i)
        aP[i] = xh + (mBase + (size_t)i * 16 + r16) * (size_t)DIM_K + half * 8;
    const _Float16* bP[4];
#pragma unroll
    for (int j = 0; j < 4; ++j)
        bP[j] = wh + (nBase + (size_t)j * 16 + r16) * (size_t)DIM_K + half * 8;

    v8f acc[4][4] = {};

    for (int k0 = 0; k0 < DIM_K; k0 += 32) {
        v16h a[4];
#pragma unroll
        for (int i = 0; i < 4; ++i)
            a[i] = load_frag(aP[i] + k0, aP[i] + k0 + 16);
        v16h b[4];
#pragma unroll
        for (int j = 0; j < 4; ++j)
            b[j] = load_frag(bP[j] + k0, bP[j] + k0 + 16);

#pragma unroll
        for (int i = 0; i < 4; ++i)
#pragma unroll
            for (int j = 0; j < 4; ++j)
                acc[i][j] = __builtin_amdgcn_wmma_f32_16x16x32_f16(
                    /*neg_a=*/false, a[i], /*neg_b=*/false, b[j],
                    /*c_mod=*/(short)0, acc[i][j],
                    /*reuse_a=*/false, /*reuse_b=*/false);
    }

    // epilogue: C/D layout — VGPR v, lane L: M = v + 8*(L>>4), N = L&15
#pragma unroll
    for (int i = 0; i < 4; ++i) {
#pragma unroll
        for (int j = 0; j < 4; ++j) {
            const size_t grow = mBase + (size_t)i * 16 + (size_t)half * 8;
            const size_t gcol = nBase + (size_t)j * 16 + r16;
            float* o = out + grow * (size_t)DIM_N + gcol;
#pragma unroll
            for (int v = 0; v < 8; ++v)
                o[(size_t)v * DIM_N] = acc[i][j][v];
        }
    }
}

// ---------------------------------------------------------------------------
// launch
// ---------------------------------------------------------------------------
extern "C" void kernel_launch(void* const* d_in, const int* in_sizes, int n_in,
                              void* d_out, int out_size, void* d_ws, size_t ws_size,
                              hipStream_t stream) {
    const float* x = (const float*)d_in[0];   // [16384, 4096] f32
    const float* w = (const float*)d_in[1];   // [4096, 4096] f32
    float* out = (float*)d_out;               // [16384, 4096] f32

    // workspace layout: wh (f16, 32 MB) | xh (f16, 128 MB)
    _Float16* wh = (_Float16*)d_ws;
    _Float16* xh = (_Float16*)((char*)d_ws + (size_t)DIM_N * DIM_K * sizeof(_Float16));

    // 1) NF4 quant->dequant of W (1024 scale blocks)
    nf4_dequant_kernel<<<dim3((DIM_N * DIM_K) / (256 * 64)), dim3(256), 0, stream>>>(w, wh);

    // 2) x f32 -> f16
    const long n4 = (long)DIM_M * DIM_K / 4;
    f32_to_f16_kernel<<<dim3(4096), dim3(256), 0, stream>>>(x, xh, n4);

    // 3) WMMA GEMM, 64x64 wave tile
    dim3 grid(DIM_N / 256, DIM_M / 128);
    gemm_f16_wmma_kernel<<<grid, dim3(256), 0, stream>>>(xh, wh, out);
}
